// MultiHeadSelfAttention_21775484190892
// MI455X (gfx1250) — matrix operand order repaired and checked by reading, not compiled, not run on hardware
//
#include <hip/hip_runtime.h>

// ---------------------------------------------------------------------------
// MHA for MI455X (gfx1250): bf16 WMMA for all three matmul stages,
// fp32 accumulate, flash-style online softmax in base-2 domain,
// row-sums on the matrix pipe (P x ones), async-LDS staging for GEMM tiles
// and for the attention K/V stream (double-buffered, s_wait_asynccnt).
// ---------------------------------------------------------------------------

typedef __bf16 bf16;
typedef __attribute__((ext_vector_type(16))) __bf16 v16bf;
typedef __attribute__((ext_vector_type(8)))  __bf16 v8bf;
typedef __attribute__((ext_vector_type(8)))  float  v8f;

#define EMB   768
#define NSEQ  4096
#define NHEAD 12
#define DH    64

// 1/sqrt(64) * log2(e): folds softmax scaling + base-2 conversion into Q.
#define QSCALE 0.1803368801111204f

#if defined(__has_builtin)
#if __has_builtin(__builtin_amdgcn_global_load_async_to_lds_b128) && \
    __has_builtin(__builtin_amdgcn_s_wait_asynccnt)
#define ASYNC_LDS 1
#endif
#endif
#ifndef ASYNC_LDS
#define ASYNC_LDS 0
#endif

static __device__ __forceinline__ v16bf cat8(v8bf lo, v8bf hi) {
  return __builtin_shufflevector(lo, hi, 0,1,2,3,4,5,6,7,8,9,10,11,12,13,14,15);
}

#if ASYNC_LDS
// Builtin params are vector_size(16) int pointers; arg0 global (AS1), arg1 LDS (AS3).
typedef int v4i_gcc __attribute__((vector_size(16)));
typedef __attribute__((address_space(1))) v4i_gcc gv4i;
typedef __attribute__((address_space(3))) v4i_gcc lv4i;
static __device__ __forceinline__ void cp16(const bf16* g, bf16* l) {
  __builtin_amdgcn_global_load_async_to_lds_b128((gv4i*)g, (lv4i*)l, 0, 0);
}
#define WAIT_ASYNC(n) __builtin_amdgcn_s_wait_asynccnt(n)
#else
static __device__ __forceinline__ void cp16(const bf16* g, bf16* l) {
  *(v8bf*)l = *(const v8bf*)g;
}
#define WAIT_ASYNC(n)
#endif

// ---------------------------------------------------------------------------
// fp32 -> bf16 conversion (sizes are multiples of 4)
// ---------------------------------------------------------------------------
__global__ void cvt_f32_bf16(const float* __restrict__ in, bf16* __restrict__ out, int n) {
  int i = (blockIdx.x * blockDim.x + threadIdx.x) * 4;
  if (i + 3 < n) {
    float4 f = *(const float4*)(in + i);
    out[i + 0] = (bf16)f.x;
    out[i + 1] = (bf16)f.y;
    out[i + 2] = (bf16)f.z;
    out[i + 3] = (bf16)f.w;
  }
}

// ---------------------------------------------------------------------------
// GEMM: C[M,N] = A[M,K] * B[N,K]^T + bias  (A,B bf16 row-major, fp32 accum)
// Block tile 128x128, 256 threads = 8 waves, wave tile 32x64 (2x4 WMMA tiles).
// Double-buffered LDS staged by GLOBAL_LOAD_ASYNC_TO_LDS_B128.
// MODE 0: scatter into Q(x QSCALE) [H][N][Dh], K [H][N][Dh], Vt [H][Dh][N]
// MODE 1: plain fp32 store to fout[M,N]
// ---------------------------------------------------------------------------
template <int MODE>
__global__ __launch_bounds__(256)
void gemm_bf16(const bf16* __restrict__ A, const bf16* __restrict__ B,
               const float* __restrict__ bias, int M, int N, int K,
               bf16* __restrict__ qout, bf16* __restrict__ kout,
               bf16* __restrict__ vtout, float* __restrict__ fout) {
  __shared__ bf16 As[2][128][40];   // 32 K cols + 8 pad (80B stride, 16B aligned)
  __shared__ bf16 Bs[2][128][40];

  const int tid  = threadIdx.x;
  const int lane = tid & 31;
  const int lr   = lane & 15;
  const int half = lane >> 4;
  const int w    = tid >> 5;
  const int wr   = w >> 1;       // 0..3 : wave row  (32 rows each)
  const int wc   = w & 1;        // 0..1 : wave col  (64 cols each)
  const int m0   = blockIdx.y * 128;
  const int n0   = blockIdx.x * 128;

  v8f acc[2][4] = {};

  const int lrow = tid >> 1;            // 0..127
  const int lch  = (tid & 1) * 16;      // 0 or 16

  auto issue_tile = [&](int buf, int kt) {
    const bf16* ap = A + (size_t)(m0 + lrow) * K + kt + lch;
    const bf16* bp = B + (size_t)(n0 + lrow) * K + kt + lch;
    cp16(ap,     &As[buf][lrow][lch]);
    cp16(ap + 8, &As[buf][lrow][lch + 8]);
    cp16(bp,     &Bs[buf][lrow][lch]);
    cp16(bp + 8, &Bs[buf][lrow][lch + 8]);
  };
  issue_tile(0, 0);

  for (int kt = 0, it = 0; kt < K; kt += 32, ++it) {
    const int cur = it & 1;
    if (kt + 32 < K) {
      issue_tile(cur ^ 1, kt + 32);
      WAIT_ASYNC(4);   // this tile's 4 loads retired (in-order completion)
    } else {
      WAIT_ASYNC(0);
    }
    __syncthreads();

    // B fragments: lane = column n; lanes<16 hold K 0..15, lanes>=16 K 16..31
    v16bf bfr[4];
#pragma unroll
    for (int j = 0; j < 4; ++j) {
      const int bn = wc * 64 + j * 16 + lr;
      const int bk = half * 16;
      bfr[j] = cat8(*(const v8bf*)&Bs[cur][bn][bk],
                    *(const v8bf*)&Bs[cur][bn][bk + 8]);
    }
    // A fragments: lane = row m; lanes<16 hold K 0..15, >=16 K 16..31
#pragma unroll
    for (int i = 0; i < 2; ++i) {
      const int am  = wr * 32 + i * 16 + lr;
      const int klo = half * 16;
      v16bf afr = cat8(*(const v8bf*)&As[cur][am][klo],
                       *(const v8bf*)&As[cur][am][klo + 8]);
#pragma unroll
      for (int j = 0; j < 4; ++j)
        acc[i][j] = __builtin_amdgcn_wmma_f32_16x16x32_bf16(
            false, afr, false, bfr[j], (short)0, acc[i][j], false, false);
    }
    __syncthreads();
  }

  // Epilogue. C layout: VGPR r -> row r (lanes 0-15) / row r+8 (lanes 16-31).
#pragma unroll
  for (int i = 0; i < 2; ++i) {
#pragma unroll
    for (int j = 0; j < 4; ++j) {
      const int gn = n0 + wc * 64 + j * 16 + lr;
      const float bv = bias[gn];
#pragma unroll
      for (int r = 0; r < 8; ++r) {
        const int gm = m0 + wr * 32 + i * 16 + r + half * 8;
        const float v = acc[i][j][r] + bv;
        if (MODE == 0) {
          const int part = gn / EMB;          // 0:Q 1:K 2:V
          const int col  = gn - part * EMB;
          const int h    = col >> 6;
          const int d    = col & 63;
          if (part == 0)
            qout[((size_t)(h * NSEQ + gm)) * DH + d] = (bf16)(v * QSCALE);
          else if (part == 1)
            kout[((size_t)(h * NSEQ + gm)) * DH + d] = (bf16)v;
          else
            vtout[((size_t)(h * DH + d)) * NSEQ + gm] = (bf16)v;  // transposed V
        } else {
          fout[(size_t)gm * N + gn] = v;
        }
      }
    }
  }
}

// ---------------------------------------------------------------------------
// Flash attention: grid (NSEQ/128, NHEAD), 256 threads = 8 waves.
// Each wave owns 16 query rows. K/V blocks (64 keys) are staged ONCE per
// workgroup into double-buffered LDS with async-to-LDS copies (8x less L2
// traffic than per-wave fragment loads), overlapped with compute.
// Q is pre-scaled by QSCALE, so scores are already in the base-2 exp domain.
// Row-sums accumulate via WMMA against a ones-matrix (every lane gets l).
// ---------------------------------------------------------------------------
__global__ __launch_bounds__(256)
void flash_attn(const bf16* __restrict__ q, const bf16* __restrict__ k,
                const bf16* __restrict__ vt, bf16* __restrict__ attnb) {
  __shared__ bf16 Ks[2][64][72];   // [key][d]   rows padded: 144B stride
  __shared__ bf16 Vs[2][64][72];   // [d][key]
  __shared__ bf16 Ps[8][16][72];   // per-wave P transpose staging

  const int tid  = threadIdx.x;
  const int lane = tid & 31;
  const int lr   = lane & 15;
  const int half = lane >> 4;
  const int w    = tid >> 5;
  const int head = blockIdx.y;
  const int q0   = blockIdx.x * 128 + w * 16;

  const bf16* Qh = q  + (size_t)head * NSEQ * DH;
  const bf16* Kh = k  + (size_t)head * NSEQ * DH;
  const bf16* Vh = vt + (size_t)head * DH * NSEQ;

  // K/V block staging: 256 threads x (2+2) x 16B = 16KB per block
  const int crow = tid >> 2;           // 0..63
  const int ccol = (tid & 3) * 16;     // 0,16,32,48
  auto issue_blk = [&](int buf, int kb) {
    const bf16* kp = Kh + (size_t)(kb + crow) * DH + ccol;
    const bf16* vp = Vh + (size_t)crow * NSEQ + kb + ccol;
    cp16(kp,     &Ks[buf][crow][ccol]);
    cp16(kp + 8, &Ks[buf][crow][ccol + 8]);
    cp16(vp,     &Vs[buf][crow][ccol]);
    cp16(vp + 8, &Vs[buf][crow][ccol + 8]);
  };

  // Q A-fragments (2 k-steps over Dh=64), resident for whole kernel
  v16bf qf[2];
#pragma unroll
  for (int s = 0; s < 2; ++s) {
    const bf16* qp = Qh + (size_t)(q0 + lr) * DH + s * 32 + half * 16;
    qf[s] = cat8(*(const v8bf*)qp, *(const v8bf*)(qp + 8));
  }

  // B-fragment of all-ones (64x16): P x ones -> every column = row sum of P
  v16bf ones;
#pragma unroll
  for (int i = 0; i < 16; ++i) ones[i] = (bf16)1.0f;

  v8f o[4] = {};
  v8f lacc = {};
  float mrow[8];
#pragma unroll
  for (int r = 0; r < 8; ++r) mrow[r] = -1e30f;

  issue_blk(0, 0);

  for (int it = 0; it < NSEQ / 64; ++it) {
    const int kb  = it * 64;
    const int cur = it & 1;
    if (kb + 64 < NSEQ) {
      issue_blk(cur ^ 1, kb + 64);
      WAIT_ASYNC(4);
    } else {
      WAIT_ASYNC(0);
    }
    __syncthreads();

    // ---- S = Q * K^T  (16 x 64 per wave), log2-domain scaled ----
    v8f sa[4] = {};
#pragma unroll
    for (int s = 0; s < 2; ++s) {
#pragma unroll
      for (int t = 0; t < 4; ++t) {
        const bf16* kp = &Ks[cur][t * 16 + lr][s * 32 + half * 16];
        v16bf bfr = cat8(*(const v8bf*)kp, *(const v8bf*)(kp + 8));
        sa[t] = __builtin_amdgcn_wmma_f32_16x16x32_bf16(
            false, qf[s], false, bfr, (short)0, sa[t], false, false);
      }
    }

    // ---- online softmax stats (xor-shuffles stay inside 16-lane half) ----
    float corr[8];
#pragma unroll
    for (int r = 0; r < 8; ++r) {
      float mx = mrow[r];
#pragma unroll
      for (int t = 0; t < 4; ++t) mx = fmaxf(mx, sa[t][r]);
#pragma unroll
      for (int d = 1; d < 16; d <<= 1) mx = fmaxf(mx, __shfl_xor(mx, d, 32));
      corr[r] = exp2f(mrow[r] - mx);
      mrow[r] = mx;
    }
    // P = exp2(S - m) -> bf16, transpose through wave-private LDS
#pragma unroll
    for (int t = 0; t < 4; ++t) {
#pragma unroll
      for (int r = 0; r < 8; ++r) {
        Ps[w][r + half * 8][t * 16 + lr] = (bf16)exp2f(sa[t][r] - mrow[r]);
      }
    }
    // rescale running O and l by correction factor
#pragma unroll
    for (int t = 0; t < 4; ++t)
#pragma unroll
      for (int r = 0; r < 8; ++r) o[t][r] *= corr[r];
#pragma unroll
    for (int r = 0; r < 8; ++r) lacc[r] *= corr[r];

    // ---- O += P * V ; l += P * ones  (P, V fragments from LDS) ----
#pragma unroll
    for (int s = 0; s < 2; ++s) {
      const int kk = s * 32 + half * 16;
      v16bf pf = cat8(*(const v8bf*)&Ps[w][lr][kk], *(const v8bf*)&Ps[w][lr][kk + 8]);
      lacc = __builtin_amdgcn_wmma_f32_16x16x32_bf16(
          false, pf, false, ones, (short)0, lacc, false, false);
#pragma unroll
      for (int t = 0; t < 4; ++t) {
        const bf16* vp = &Vs[cur][t * 16 + lr][s * 32 + half * 16];
        v16bf bfr = cat8(*(const v8bf*)vp, *(const v8bf*)(vp + 8));
        o[t] = __builtin_amdgcn_wmma_f32_16x16x32_bf16(
            false, pf, false, bfr, (short)0, o[t], false, false);
      }
    }
    __syncthreads();   // all reads of buffer `cur` done before it is refilled
  }

  // ---- epilogue: normalize, store bf16 head-concatenated ----
  float linv[8];
#pragma unroll
  for (int r = 0; r < 8; ++r) linv[r] = 1.f / lacc[r];
#pragma unroll
  for (int t = 0; t < 4; ++t) {
#pragma unroll
    for (int r = 0; r < 8; ++r) {
      const int gm = q0 + r + half * 8;
      const int gn = head * DH + t * 16 + lr;
      attnb[(size_t)gm * EMB + gn] = (bf16)(o[t][r] * linv[r]);
    }
  }
}

// ---------------------------------------------------------------------------
extern "C" void kernel_launch(void* const* d_in, const int* in_sizes, int n_in,
                              void* d_out, int out_size, void* d_ws, size_t ws_size,
                              hipStream_t stream) {
  const float* x     = (const float*)d_in[0];
  const float* w_qkv = (const float*)d_in[1];
  const float* b_qkv = (const float*)d_in[2];
  const float* w_out = (const float*)d_in[3];
  const float* b_out = (const float*)d_in[4];

  char* ws = (char*)d_ws;
  size_t off = 0;
  auto alloc = [&](size_t bytes) -> void* {
    void* p = ws + off;
    off += (bytes + 255) & ~(size_t)255;
    return p;
  };
  bf16* xb  = (bf16*)alloc((size_t)NSEQ * EMB * 2);
  bf16* wqb = (bf16*)alloc((size_t)3 * EMB * EMB * 2);
  bf16* wob = (bf16*)alloc((size_t)EMB * EMB * 2);
  bf16* qb  = (bf16*)alloc((size_t)NSEQ * EMB * 2);
  bf16* kb  = (bf16*)alloc((size_t)NSEQ * EMB * 2);
  bf16* vtb = (bf16*)alloc((size_t)NSEQ * EMB * 2);
  bf16* ab  = (bf16*)alloc((size_t)NSEQ * EMB * 2);

  const int nx  = NSEQ * EMB;
  const int nwq = 3 * EMB * EMB;
  const int nwo = EMB * EMB;
  cvt_f32_bf16<<<(nx / 4 + 255) / 256, 256, 0, stream>>>(x, xb, nx);
  cvt_f32_bf16<<<(nwq / 4 + 255) / 256, 256, 0, stream>>>(w_qkv, wqb, nwq);
  cvt_f32_bf16<<<(nwo / 4 + 255) / 256, 256, 0, stream>>>(w_out, wob, nwo);

  dim3 g1(3 * EMB / 128, NSEQ / 128);
  gemm_bf16<0><<<g1, 256, 0, stream>>>(xb, wqb, b_qkv, NSEQ, 3 * EMB, EMB,
                                       qb, kb, vtb, nullptr);

  dim3 g2(NSEQ / 128, NHEAD);
  flash_attn<<<g2, 256, 0, stream>>>(qb, kb, vtb, ab);

  dim3 g3(EMB / 128, NSEQ / 128);
  gemm_bf16<1><<<g3, 256, 0, stream>>>(ab, wob, b_out, NSEQ, EMB, EMB,
                                       nullptr, nullptr, nullptr, (float*)d_out);
}
